// AttentionModule_4922032521654
// MI455X (gfx1250) — matrix operand order repaired and checked by reading, not compiled, not run on hardware
//
#include <hip/hip_runtime.h>
#include <hip/hip_bf16.h>

typedef __bf16 bf16;
typedef __bf16 v16bf  __attribute__((ext_vector_type(16)));
typedef __bf16 bf16x8 __attribute__((ext_vector_type(8)));
typedef float  v8f    __attribute__((ext_vector_type(8)));

#define NC   128
#define W    64
#define H    64
#define L    4096      // W*H
#define KDIM 1152      // NC*3*3

// ---------------------------------------------------------------------------
// Stage 1: build bf16 operand matrices.
//   kflat [L][KDIM]  = normalized (bg-patch + 1e-7), row-major (B^T for GEMM1)
//   kflatT[KDIM][L]  = transpose of kflat              (B^T for GEMM2)
//   fpatch[L][KDIM]  = raw feat patches                (A for GEMM1)
// One block per patch l, 256 threads over the 1152 patch elements.
// ---------------------------------------------------------------------------
__global__ __launch_bounds__(256) void prep_kernel(
    const float* __restrict__ feat, const float* __restrict__ mask,
    bf16* __restrict__ kflat, bf16* __restrict__ kflatT, bf16* __restrict__ fpatch)
{
    __shared__ float kv[KDIM];
    __shared__ float fv[KDIM];
    __shared__ float red[256];

    const int l = blockIdx.x;
    const int y = l >> 6;
    const int x = l & 63;

    float ssq = 0.f;
    for (int k = threadIdx.x; k < KDIM; k += 256) {
        int c  = k / 9;
        int r  = k - c * 9;
        int dy = r / 3, dx = r % 3;
        int yy = y + dy - 1, xx = x + dx - 1;
        float f = 0.f, m = 0.f;
        if (yy >= 0 && yy < W && xx >= 0 && xx < H) {
            f = feat[(size_t)c * L + yy * H + xx];
            m = mask[yy * H + xx];
        }
        float kvv = f * m + 1e-7f;   // reference adds 1e-7 before the norm
        kv[k] = kvv;
        fv[k] = f;
        ssq += kvv * kvv;
    }
    red[threadIdx.x] = ssq;
    __syncthreads();
    for (int s = 128; s > 0; s >>= 1) {
        if (threadIdx.x < s) red[threadIdx.x] += red[threadIdx.x + s];
        __syncthreads();
    }
    const float inv = rsqrtf(red[0]);

    for (int k = threadIdx.x; k < KDIM; k += 256) {
        bf16 kn = (bf16)(kv[k] * inv);
        kflat [(size_t)l * KDIM + k] = kn;
        kflatT[(size_t)k * L    + l] = kn;
        fpatch[(size_t)l * KDIM + k] = (bf16)fv[k];
    }
}

// ---------------------------------------------------------------------------
// Stage 2/4: register-blocked WMMA GEMM.
//   C[M][N] (f32) = A[M][K] (bf16, row-major, stride lda)
//                 x Bt[N][K]^T (bf16, row-major, stride ldbt)
// Block tile 128x128; 4 waves arranged 2(M) x 2(N); wave tile 64x64
// = 4x4 fragments of v_wmma_f32_16x16x32_bf16 -> 16 WMMA per 16 b128 loads
// per K-step, with 4x register reuse of every fragment.
// A/B fragments use contiguous k (k = 16*(lane>>4) + s) for both; C/D layout per ISA (wave32).
// ---------------------------------------------------------------------------
__global__ __launch_bounds__(128) void gemm_bf16f32(
    const bf16* __restrict__ A, const bf16* __restrict__ Bt, float* __restrict__ C,
    int K, int lda, int ldbt, int ldc)
{
    const int lane = threadIdx.x & 31;
    const int wave = threadIdx.x >> 5;       // 0..3
    const int wm   = wave >> 1;              // 0..1
    const int wn   = wave & 1;               // 0..1
    const int rowBase = blockIdx.y * 128 + wm * 64;
    const int colBase = blockIdx.x * 128 + wn * 64;

    // A fragment (16x32, 16-bit): lane<16 -> K 0..15, lane>=16 -> K 16..31
    const int am  = lane & 15;
    const int akh = (lane >> 4) << 4;        // 0 or 16
    // B fragment (32x16, 16-bit): lane&15 = column, lane>>4 selects K-half of 16
    const int bn  = lane & 15;
    const int bkh = (lane >> 4) << 4;        // 0 or 16

    v8f acc[4][4] = {};

    union U { v16bf v; bf16x8 h[2]; };

    const bf16* Arow[4];
    const bf16* Brow[4];
    #pragma unroll
    for (int mi = 0; mi < 4; mi++)
        Arow[mi] = A + (size_t)(rowBase + mi * 16 + am) * lda + akh;
    #pragma unroll
    for (int ni = 0; ni < 4; ni++)
        Brow[ni] = Bt + (size_t)(colBase + ni * 16 + bn) * ldbt + bkh;

    for (int k0 = 0; k0 < K; k0 += 32) {
        U a[4];
        #pragma unroll
        for (int mi = 0; mi < 4; mi++) {
            const bf16x8* pa = (const bf16x8*)(Arow[mi] + k0);
            a[mi].h[0] = pa[0];   // K = k0+akh    .. +7
            a[mi].h[1] = pa[1];   // K = k0+akh+8  .. +15
        }
        U b[4];
        #pragma unroll
        for (int ni = 0; ni < 4; ni++) {
            const bf16x8* pb = (const bf16x8*)(Brow[ni] + k0);
            b[ni].h[0] = pb[0];   // K = k0+bkh    .. +7
            b[ni].h[1] = pb[1];   // K = k0+bkh+8  .. +15
        }
        // pull the next K-tile toward the WGP while the XDL pipe works
        if (k0 + 32 < K) {
            __builtin_prefetch(Arow[0] + k0 + 32, 0, 1);
            __builtin_prefetch(Brow[0] + k0 + 32, 0, 1);
        }
        #pragma unroll
        for (int mi = 0; mi < 4; mi++)
            #pragma unroll
            for (int ni = 0; ni < 4; ni++)
                acc[mi][ni] = __builtin_amdgcn_wmma_f32_16x16x32_bf16(
                    false, a[mi].v, false, b[ni].v,
                    (short)0, acc[mi][ni], false, false);
    }

    // C/D layout: VGPR r -> M = r + (lane>>4)*8, N = lane&15
    #pragma unroll
    for (int mi = 0; mi < 4; mi++) {
        #pragma unroll
        for (int ni = 0; ni < 4; ni++) {
            int r0 = rowBase + mi * 16 + ((lane >> 4) << 3);
            int c0 = colBase + ni * 16 + (lane & 15);
            #pragma unroll
            for (int r = 0; r < 8; r++)
                C[(size_t)(r0 + r) * ldc + c0] = acc[mi][ni][r];
        }
    }
}

// ---------------------------------------------------------------------------
// Stage 3: fused 3x3 score propagation (SAME, zero-pad) + softmax over l,
// output as bf16 attention (A operand of GEMM2).
// One block per spatial position p; 4096-wide row staged in LDS (16 KB).
// ---------------------------------------------------------------------------
__global__ __launch_bounds__(256) void prop_softmax_kernel(
    const float* __restrict__ S, bf16* __restrict__ attn)
{
    __shared__ float prop[L];
    __shared__ float red[256];

    const int p = blockIdx.x;
    const int y = p >> 6;
    const int x = p & 63;

    float lmax = -1e30f;
    for (int l = threadIdx.x; l < L; l += 256) {
        float s = 0.f;
        #pragma unroll
        for (int dy = -1; dy <= 1; dy++) {
            int yy = y + dy;
            if (yy < 0 || yy >= W) continue;
            #pragma unroll
            for (int dx = -1; dx <= 1; dx++) {
                int xx = x + dx;
                if (xx < 0 || xx >= H) continue;
                s += S[(size_t)(yy * H + xx) * L + l];
            }
        }
        prop[l] = s;
        lmax = fmaxf(lmax, s);
    }
    red[threadIdx.x] = lmax;
    __syncthreads();
    for (int s2 = 128; s2 > 0; s2 >>= 1) {
        if (threadIdx.x < s2)
            red[threadIdx.x] = fmaxf(red[threadIdx.x], red[threadIdx.x + s2]);
        __syncthreads();
    }
    const float mx = red[0];
    __syncthreads();

    float lsum = 0.f;
    for (int l = threadIdx.x; l < L; l += 256) {
        float e = __expf(prop[l] - mx);
        prop[l] = e;
        lsum += e;
    }
    red[threadIdx.x] = lsum;
    __syncthreads();
    for (int s2 = 128; s2 > 0; s2 >>= 1) {
        if (threadIdx.x < s2) red[threadIdx.x] += red[threadIdx.x + s2];
        __syncthreads();
    }
    const float inv = 1.f / red[0];

    for (int l = threadIdx.x; l < L; l += 256)
        attn[(size_t)p * L + l] = (bf16)(prop[l] * inv);
}

// ---------------------------------------------------------------------------
// Stage 5: overlap-add of weighted patches + mask blend.
// recovered[c,Y,X] = sum_{dy,dx} mout[(Y+1-dy, X+1-dx)][c,dy,dx]
// out = recovered*(1-mask)/9 + feat*mask
// ---------------------------------------------------------------------------
__global__ __launch_bounds__(256) void output_kernel(
    const float* __restrict__ mout, const float* __restrict__ feat,
    const float* __restrict__ mask, float* __restrict__ out)
{
    int idx = blockIdx.x * 256 + threadIdx.x;     // over NC*L
    if (idx >= NC * L) return;
    int c   = idx >> 12;
    int pos = idx & (L - 1);
    int Y = pos >> 6, X = pos & 63;

    float acc = 0.f;
    #pragma unroll
    for (int dy = 0; dy < 3; dy++) {
        int y = Y + 1 - dy;
        if (y < 0 || y >= W) continue;
        #pragma unroll
        for (int dx = 0; dx < 3; dx++) {
            int x = X + 1 - dx;
            if (x < 0 || x >= H) continue;
            acc += mout[(size_t)(y * H + x) * KDIM + (c * 3 + dy) * 3 + dx];
        }
    }
    float m = mask[pos];
    float f = feat[(size_t)c * L + pos];
    out[(size_t)c * L + pos] = acc * (1.f - m) * (1.f / 9.f) + f * m;
}

// ---------------------------------------------------------------------------
extern "C" void kernel_launch(void* const* d_in, const int* in_sizes, int n_in,
                              void* d_out, int out_size, void* d_ws, size_t ws_size,
                              hipStream_t stream)
{
    const float* feat_all = (const float*)d_in[0];   // (2,128,64,64) f32
    const float* mask_all = (const float*)d_in[1];   // (2,1,64,64)   f32
    float* out = (float*)d_out;                      // (2,128,64,64) f32

    char* ws = (char*)d_ws;
    size_t off = 0;
    bf16*  kflat  = (bf16*) (ws + off); off += (size_t)L * KDIM * 2;   //  9.4 MB
    bf16*  kflatT = (bf16*) (ws + off); off += (size_t)KDIM * L * 2;   //  9.4 MB
    bf16*  fpatch = (bf16*) (ws + off); off += (size_t)L * KDIM * 2;   //  9.4 MB
    float* scores = (float*)(ws + off); off += (size_t)L * L * 4;      // 64.0 MB
    bf16*  attn   = (bf16*) (ws + off); off += (size_t)L * L * 2;      // 33.6 MB
    float* mout   = (float*)(ws + off); off += (size_t)L * KDIM * 4;   // 18.9 MB

    for (int b = 0; b < 2; b++) {
        const float* feat = feat_all + (size_t)b * NC * L;
        const float* mask = mask_all + (size_t)b * L;

        prep_kernel<<<L, 256, 0, stream>>>(feat, mask, kflat, kflatT, fpatch);

        // scores[p][l] = fpatch[p] . kflat[l]   (M=N=4096, K=1152)
        gemm_bf16f32<<<dim3(L / 128, L / 128), 128, 0, stream>>>(
            fpatch, kflat, scores, KDIM, KDIM, KDIM, L);

        prop_softmax_kernel<<<L, 256, 0, stream>>>(scores, attn);

        // mout[p][k] = sum_l attn[p][l] * kflat[l][k]  (M=4096, N=1152, K=4096)
        gemm_bf16f32<<<dim3(KDIM / 128, L / 128), 128, 0, stream>>>(
            attn, kflatT, mout, L, L, L, KDIM);

        output_kernel<<<(NC * L + 255) / 256, 256, 0, stream>>>(
            mout, feat, mask, out + (size_t)b * NC * L);
    }
}
